// VectorQuantizer_88819923681653
// MI455X (gfx1250) — matrix operand order repaired and checked by reading, not compiled, not run on hardware
//
#include <hip/hip_runtime.h>

typedef __attribute__((ext_vector_type(16))) _Float16 v16h;
typedef __attribute__((ext_vector_type(8)))  float    v8f;
typedef __attribute__((ext_vector_type(2)))  float    v2f;

#define VOCAB 1024
#define EMBED 256
#define SEQ   2048
#define NROWS 65536

static constexpr size_t ZQ_OFF   = 1ull;
static constexpr size_t PERP_OFF = 1ull + 16777216ull;          // 16777217
static constexpr size_t OH_OFF   = PERP_OFF + 1ull;             // 16777218
static constexpr size_t IDX_OFF  = OH_OFF + 67108864ull;        // 83886082

// workspace byte offsets
static constexpr size_t WS_EMBSW = 0ull;        // 262144 _Float16 (512 KB), WMMA-B swizzled
static constexpr size_t WS_NORM  = 524288ull;   // 1024 float  ||e||^2
static constexpr size_t WS_HIST  = 528384ull;   // 1024 int    histogram
static constexpr size_t WS_LOSSP = 532480ull;   // 512 float   per-block loss partials

// Build emb in f16, laid out as [ct][kb][lane][16] so a B-fragment load is one
// contiguous 32B per lane: element (ct,kb,lane,h) = emb[ct*16+lane%16][kb*32+(lane>=16)*16+h]
__global__ __launch_bounds__(512) void vq_prep_emb(const float* __restrict__ emb,
                                                   _Float16* __restrict__ embSw) {
    int i    = blockIdx.x * 512 + threadIdx.x;   // 0..262143
    int h    = i & 15;
    int lane = (i >> 4) & 31;
    int kb   = (i >> 9) & 7;
    int ct   = i >> 12;
    int code = ct * 16 + (lane & 15);
    int k    = kb * 32 + ((lane >> 4) << 4) + h;
    embSw[i] = (_Float16)emb[code * EMBED + k];
}

__global__ __launch_bounds__(256) void vq_prep_norm(const float* __restrict__ emb,
                                                    float* __restrict__ embNorm,
                                                    int* __restrict__ hist) {
    __shared__ float sr[256];
    int code = blockIdx.x, t = threadIdx.x;
    float v = emb[code * EMBED + t];
    sr[t] = v * v;
    __syncthreads();
    for (int s = 128; s > 0; s >>= 1) { if (t < s) sr[t] += sr[t + s]; __syncthreads(); }
    if (t == 0) { embNorm[code] = sr[0]; hist[code] = 0; }
}

__global__ __launch_bounds__(128) void vq_main(const float* __restrict__ z,
                                               const float* __restrict__ emb32,
                                               const _Float16* __restrict__ embSw,
                                               const float* __restrict__ embNorm,
                                               int* __restrict__ hist,
                                               float* __restrict__ lossPartial,
                                               float* __restrict__ out) {
    __shared__ __align__(32) _Float16 zA[8][8][32][16];   // 64 KB: 8 M-tiles of 16 rows
    __shared__ __align__(32) _Float16 eB[2][8][32][16];   // 16 KB: double-buffered col tile
    __shared__ int   idxArr[8][16];
    __shared__ float red[128];

    const int tid  = threadIdx.x;
    const int w    = tid >> 5;
    const int lane = tid & 31;
    const int n0   = blockIdx.x * 128;         // 128 rows per block; never crosses batch b
    const int b    = n0 / SEQ;
    const int s0   = n0 % SEQ;

    // ---- stage z tile: thread = row; f32 -> f16 swizzled A fragments; fp32 sum(z^2) ----
    float zsq = 0.f;
    {
        const int r  = tid;                    // row 0..127
        const int ww = r >> 4, M = r & 15;
        const size_t base = ((size_t)b * EMBED) * SEQ + (size_t)s0 + r;
#pragma unroll 4
        for (int d = 0; d < EMBED; ++d) {      // 512B coalesced per d across the block
            float v = z[base + (size_t)d * SEQ];
            zsq += v * v;
            int kb = d >> 5, j = d & 31;
            int ln  = M + ((j >> 4) << 4);
            int pos = j & 15;
            zA[ww][kb][ln][pos] = (_Float16)v;
        }
    }
    __syncthreads();

    // A fragments: wave w owns M-tiles 2w and 2w+1 (rows n0+32w .. n0+32w+31)
    v16h a0[8], a1[8];
#pragma unroll
    for (int kb = 0; kb < 8; ++kb) {
        a0[kb] = *(const v16h*)&zA[2 * w + 0][kb][lane][0];
        a1[kb] = *(const v16h*)&zA[2 * w + 1][kb][lane][0];
    }

    float minv0[8], minv1[8];
    int   mini0[8], mini1[8];
#pragma unroll
    for (int i = 0; i < 8; ++i) {
        minv0[i] = 3.4e38f; mini0[i] = 0;
        minv1[i] = 3.4e38f; mini1[i] = 0;
    }

    // async double-buffered emb column-tile staging (8 KB per tile)
    const unsigned ldsE0 = (unsigned)(uintptr_t)&eB[0][0][0][0] + (unsigned)tid * 16u;
    const unsigned ldsE1 = (unsigned)(uintptr_t)&eB[1][0][0][0] + (unsigned)tid * 16u;
    const char* egl = (const char*)embSw;

#define VQ_ISSUE_ASYNC(CT, LDSA)                                                  \
    do {                                                                          \
        unsigned go = (unsigned)((CT) * 8192 + tid * 16);                         \
        asm volatile(                                                             \
            "global_load_async_to_lds_b128 %0, %1, %2 offset:0\n\t"               \
            "global_load_async_to_lds_b128 %0, %1, %2 offset:2048\n\t"            \
            "global_load_async_to_lds_b128 %0, %1, %2 offset:4096\n\t"            \
            "global_load_async_to_lds_b128 %0, %1, %2 offset:6144"                \
            :: "v"(LDSA), "v"(go), "s"(egl) : "memory");                          \
    } while (0)

    VQ_ISSUE_ASYNC(0, ldsE0);
    asm volatile("s_wait_asynccnt 0x0" ::: "memory");
    __syncthreads();

    for (int ct = 0; ct < 64; ++ct) {
        const int cur = ct & 1;
        if (ct < 63) {                          // prefetch next tile into the other buffer
            if (cur) VQ_ISSUE_ASYNC(ct + 1, ldsE0);
            else     VQ_ISSUE_ASYNC(ct + 1, ldsE1);
        }

        float en = embNorm[ct * 16 + (lane & 15)];
        v8f acc0 = {0.f, 0.f, 0.f, 0.f, 0.f, 0.f, 0.f, 0.f};
        v8f acc1 = {0.f, 0.f, 0.f, 0.f, 0.f, 0.f, 0.f, 0.f};
#pragma unroll
        for (int kb = 0; kb < 8; ++kb) {
            v16h bf = *(const v16h*)&eB[cur][kb][lane][0];
            acc0 = __builtin_amdgcn_wmma_f32_16x16x32_f16(false, a0[kb], false, bf,
                                                          (short)0, acc0, false, false);
            acc1 = __builtin_amdgcn_wmma_f32_16x16x32_f16(false, a1[kb], false, bf,
                                                          (short)0, acc1, false, false);
        }
        int cand = ct * 16 + (lane & 15);
#pragma unroll
        for (int i = 0; i < 8; ++i) {
            float sa = en - 2.0f * acc0[i];     // ||e||^2 - 2 z.e  (||z||^2 const per row)
            float sb = en - 2.0f * acc1[i];
            if (sa < minv0[i]) { minv0[i] = sa; mini0[i] = cand; }
            if (sb < minv1[i]) { minv1[i] = sb; mini1[i] = cand; }
        }

        asm volatile("s_wait_asynccnt 0x0" ::: "memory");
        __syncthreads();                        // next-tile data landed; buffer reuse safe
    }
#undef VQ_ISSUE_ASYNC

    // argmin across the 16 lanes of each half-wave (N dimension); first-index tie-break
#pragma unroll
    for (int off = 1; off < 16; off <<= 1) {
#pragma unroll
        for (int i = 0; i < 8; ++i) {
            float ov; int oi;
            ov = __shfl_xor(minv0[i], off, 32); oi = __shfl_xor(mini0[i], off, 32);
            if (ov < minv0[i] || (ov == minv0[i] && oi < mini0[i])) { minv0[i] = ov; mini0[i] = oi; }
            ov = __shfl_xor(minv1[i], off, 32); oi = __shfl_xor(mini1[i], off, 32);
            if (ov < minv1[i] || (ov == minv1[i] && oi < mini1[i])) { minv1[i] = ov; mini1[i] = oi; }
        }
    }

    float lossLoc = zsq;
    if (lane == 0 || lane == 16) {              // lane0: rows M=0..7, lane16: rows M=8..15
        int rbase = (lane >> 4) * 8;
#pragma unroll
        for (int i = 0; i < 8; ++i) {
            int c0 = mini0[i], c1 = mini1[i];
            int row0 = n0 + w * 32 + rbase + i;
            int row1 = row0 + 16;
            idxArr[2 * w + 0][rbase + i] = c0;
            idxArr[2 * w + 1][rbase + i] = c1;
            out[IDX_OFF + row0] = (float)c0;
            out[IDX_OFF + row1] = (float)c1;
            atomicAdd(&hist[c0], 1);
            atomicAdd(&hist[c1], 1);
            lossLoc += minv0[i] + minv1[i];     // sum(e-z)^2 = sum z^2 + sum minScore
        }
    }
    red[tid] = lossLoc;
    __syncthreads();
    for (int s2 = 64; s2 > 0; s2 >>= 1) { if (tid < s2) red[tid] += red[tid + s2]; __syncthreads(); }
    if (tid == 0) lossPartial[blockIdx.x] = red[0];

    // z_q_st rows = fp32 emb rows of the chosen codes (z_q_st == z_q numerically); NT stream
#pragma unroll 1
    for (int j = 0; j < 32; ++j) {
        int c = idxArr[2 * w + (j >> 4)][j & 15];
        size_t obase = ZQ_OFF + (size_t)(n0 + w * 32 + j) * EMBED;
        const float* e = emb32 + (size_t)c * EMBED;
#pragma unroll
        for (int k = 0; k < 8; ++k) {
            int col = k * 32 + lane;            // 128B-contiguous per instruction
            __builtin_nontemporal_store(e[col], &out[obase + col]);
        }
    }
}

__global__ __launch_bounds__(256) void vq_onehot(const float* __restrict__ idxF,
                                                 float* __restrict__ oh) {
    int n = blockIdx.x;
    int t = threadIdx.x;
    int c = (int)(idxF[n] + 0.5f);
    size_t base = (size_t)n * VOCAB + (size_t)t * 4;   // 8B aligned (OH_OFF is even)
    v2f lo, hi;
    lo.x = (t * 4 + 0 == c) ? 1.f : 0.f;
    lo.y = (t * 4 + 1 == c) ? 1.f : 0.f;
    hi.x = (t * 4 + 2 == c) ? 1.f : 0.f;
    hi.y = (t * 4 + 3 == c) ? 1.f : 0.f;
    __builtin_nontemporal_store(lo, (v2f*)(oh + base));      // 256 MB stream: keep out of L2
    __builtin_nontemporal_store(hi, (v2f*)(oh + base + 2));
}

__global__ __launch_bounds__(1024) void vq_finalize(const int* __restrict__ hist,
                                                    const float* __restrict__ lossPartial,
                                                    float* __restrict__ out) {
    __shared__ float sl[1024];
    __shared__ float se[1024];
    int t = threadIdx.x;
    float em = (float)hist[t] * (1.0f / 65536.0f);
    se[t] = em * logf(em + 1e-10f);
    sl[t] = (t < 512) ? lossPartial[t] : 0.f;
    __syncthreads();
    for (int s = 512; s > 0; s >>= 1) {
        if (t < s) { sl[t] += sl[t + s]; se[t] += se[t + s]; }
        __syncthreads();
    }
    if (t == 0) {
        out[0]        = 1.25f * sl[0] / 16777216.0f;   // (1 + BETA) * mean
        out[PERP_OFF] = expf(-se[0]);
    }
}

extern "C" void kernel_launch(void* const* d_in, const int* in_sizes, int n_in,
                              void* d_out, int out_size, void* d_ws, size_t ws_size,
                              hipStream_t stream) {
    const float* z   = (const float*)d_in[0];   // (32, 256, 2048)
    const float* emb = (const float*)d_in[1];   // (1024, 256)
    float* out = (float*)d_out;
    char*  ws  = (char*)d_ws;

    _Float16* embSw  = (_Float16*)(ws + WS_EMBSW);
    float*    embNorm= (float*)(ws + WS_NORM);
    int*      hist   = (int*)(ws + WS_HIST);
    float*    lossP  = (float*)(ws + WS_LOSSP);

    vq_prep_emb <<<512, 512, 0, stream>>>(emb, embSw);
    vq_prep_norm<<<1024, 256, 0, stream>>>(emb, embNorm, hist);
    vq_main     <<<NROWS / 128, 128, 0, stream>>>(z, emb, embSw, embNorm, hist, lossP, out);
    vq_onehot   <<<NROWS, 256, 0, stream>>>(out + IDX_OFF, out + OH_OFF);
    vq_finalize <<<1, 1024, 0, stream>>>(hist, lossP, out);
}
